// ScaledDotProductAttention_43482248904867
// MI455X (gfx1250) — compile-verified, matrix operand order checked by reading, not run on hardware
//
#include <hip/hip_runtime.h>
#include <hip/hip_bf16.h>
#include <math.h>

// B=16, LQ=LK=2048, D=64, temperature=8
#define B_   16
#define LQ_  2048
#define LK_  2048
#define D_   64
#define TEMP_INV 0.125f
#define SD   2051   // padded LDS row stride in dwords: 2051 % 64 = 3 (odd) -> no bank conflicts
#define LOG2E 1.44269504088896340736f

typedef __attribute__((ext_vector_type(16))) __bf16 v16bf;
typedef __attribute__((ext_vector_type(8)))  float  v8f;
typedef __attribute__((ext_vector_type(4)))  unsigned int u32x4;
typedef __attribute__((ext_vector_type(8)))  int          i32x8;
typedef __attribute__((ext_vector_type(4)))  int          i32x4;

// K-index inside a 16x32 (16-bit) WMMA A/B fragment for element e of the v16bf,
// per CDNA5 ISA 7.12.2: VGPR g holds K = (g<4?0:16) + 8*(lane>=16) + 2*(g&3) + h
__device__ __forceinline__ int kof(int e, int half) {
  int g = e >> 1, h = e & 1;
  return ((g < 4) ? 0 : 16) + half * 8 + 2 * (g & 3) + h;
}

// ---------------- pass 1: partial sum/max of time_diff ----------------
__global__ void td_reduce_partial(const float* __restrict__ td,
                                  float* __restrict__ psum,
                                  float* __restrict__ pmax,
                                  long long n) {
  __shared__ float ssum[256];
  __shared__ float smax[256];
  long long stride = (long long)gridDim.x * blockDim.x;
  long long i = (long long)blockIdx.x * blockDim.x + threadIdx.x;
  float s = 0.f, m = 0.f;  // time_diff > 0
  for (; i < n; i += stride) { float x = td[i]; s += x; m = fmaxf(m, x); }
  ssum[threadIdx.x] = s; smax[threadIdx.x] = m;
  __syncthreads();
  for (int off = 128; off > 0; off >>= 1) {
    if (threadIdx.x < off) {
      ssum[threadIdx.x] += ssum[threadIdx.x + off];
      smax[threadIdx.x] = fmaxf(smax[threadIdx.x], smax[threadIdx.x + off]);
    }
    __syncthreads();
  }
  if (threadIdx.x == 0) { psum[blockIdx.x] = ssum[0]; pmax[blockIdx.x] = smax[0]; }
}

// ---------------- pass 2: finalize scalars ----------------
// scal[0] = 1/mean(td), scal[1] = softplus(tpw) * max(td)/mean(td)
__global__ void td_finalize(const float* __restrict__ psum,
                            const float* __restrict__ pmax,
                            int nblocks, const float* __restrict__ tpw,
                            float* __restrict__ scal, float n_elems) {
  __shared__ float ssum[256];
  __shared__ float smax[256];
  float s = 0.f, m = 0.f;
  for (int i = threadIdx.x; i < nblocks; i += 256) { s += psum[i]; m = fmaxf(m, pmax[i]); }
  ssum[threadIdx.x] = s; smax[threadIdx.x] = m;
  __syncthreads();
  for (int off = 128; off > 0; off >>= 1) {
    if (threadIdx.x < off) {
      ssum[threadIdx.x] += ssum[threadIdx.x + off];
      smax[threadIdx.x] = fmaxf(smax[threadIdx.x], smax[threadIdx.x + off]);
    }
    __syncthreads();
  }
  if (threadIdx.x == 0) {
    float inv_mean = n_elems / ssum[0];
    float x = tpw[0];
    float sp = (x > 0.f) ? (x + log1pf(__expf(-x))) : log1pf(__expf(x));
    scal[0] = inv_mean;
    scal[1] = sp * (smax[0] * inv_mean);
  }
}

// ---------------- fused attention ----------------
// One workgroup (8 wave32) per (batch, 16-row q tile). Score tile 16x2051 f32
// lives entirely in LDS (131 KB of the 320 KB/WGP), padded stride kills bank
// conflicts for the transposed fragment gathers in the P*V pass.
__launch_bounds__(256, 1)
__global__ void attn_fused(const float* __restrict__ q, const float* __restrict__ k,
                           const float* __restrict__ v, const float* __restrict__ td,
                           const unsigned char* __restrict__ mask,
                           const float* __restrict__ tmw_p,
                           const float* __restrict__ scal,
                           float* __restrict__ out, float* __restrict__ attn_out) {
  extern __shared__ float smem[];
  float* s_tile = smem;            // 16 * SD floats (scores -> probs)
  float* o_scr  = smem + 16 * SD;  // 8 * 256 floats (partial O tiles, then final O)

  const int tid  = threadIdx.x;
  const int lane = tid & 31;
  const int wid  = tid >> 5;
  const int half = lane >> 4;
  const int l16  = lane & 15;

  const int bq = blockIdx.x;       // 0..2047
  const int b  = bq >> 7;          // / (LQ/16)
  const int q0 = (bq & 127) << 4;

  const float inv_mean = scal[0];
  const float addc     = scal[1];
  const float k2       = tmw_p[0] * LOG2E;   // tmw / ln(2):  tmw/ln(x) = k2 * rcp(log2(x))

  // A fragments of Q tile (16x64 split into two 16x32), kept in VGPRs all kernel
  const float* qb = q + ((size_t)b * LQ_ + q0 + l16) * D_;
  v16bf aq0, aq1;
#pragma unroll
  for (int e = 0; e < 16; ++e) {
    int kk = kof(e, half);
    aq0[e] = (__bf16)qb[kk];
    aq1[e] = (__bf16)qb[kk + 32];
  }

  // ---- Phase 1: S = Q K^T / T + bias(time_diff), masked -> LDS ----
  const size_t td_base = ((size_t)b * LQ_ + q0) * (size_t)LK_;
  for (int t = wid; t < (LK_ / 16); t += 8) {
    const int n0 = t << 4;
    const float* kb = k + ((size_t)b * LK_ + n0 + l16) * D_;
    v16bf bk0, bk1;
#pragma unroll
    for (int e = 0; e < 16; ++e) {
      int kk = kof(e, half);
      bk0[e] = (__bf16)kb[kk];
      bk1[e] = (__bf16)kb[kk + 32];
    }
    if (t + 8 < (LK_ / 16)) {  // gfx1250 global_prefetch_b8 of next K tile / td tile
      __builtin_prefetch(k + ((size_t)b * LK_ + ((t + 8) << 4) + l16) * D_, 0, 0);
      __builtin_prefetch(td + td_base + ((t + 8) << 4) + l16, 0, 0);
    }

    v8f c = {};
    c = __builtin_amdgcn_wmma_f32_16x16x32_bf16(false, aq0, false, bk0, (short)0, c, false, false);
    c = __builtin_amdgcn_wmma_f32_16x16x32_bf16(false, aq1, false, bk1, (short)0, c, false, false);

#pragma unroll
    for (int g = 0; g < 8; ++g) {
      const int m   = g + 8 * half;
      const int col = n0 + l16;
      const size_t idx = td_base + (size_t)m * LK_ + col;
      float tdv = fmaf(td[idx], inv_mean, addc) + 2.71828182845904523536f;
      // c/8 + tmw/ln(e+tdv)  ==  fma(k2, rcp(log2(e+tdv)), c*0.125)
      float w   = __builtin_amdgcn_rcpf(__builtin_amdgcn_logf(tdv));
      float sc  = fmaf(k2, w, c[g] * TEMP_INV);
      if (mask[idx]) sc = -1e10f;
      s_tile[m * SD + col] = sc;
    }
  }
  __syncthreads();

  // ---- Phase 2: row softmax in LDS, stream normalized attn to global ----
  float* attn_base = attn_out + td_base;
  for (int r = wid * 2; r < wid * 2 + 2; ++r) {
    float* row = s_tile + r * SD;
    float mx = -3.4e38f;
    for (int j = lane; j < LK_; j += 32) mx = fmaxf(mx, row[j]);
#pragma unroll
    for (int off = 16; off > 0; off >>= 1) mx = fmaxf(mx, __shfl_xor(mx, off, 32));
    float sum = 0.f;
    for (int j = lane; j < LK_; j += 32) {
      float e = __builtin_amdgcn_exp2f((row[j] - mx) * LOG2E);
      row[j] = e; sum += e;
    }
#pragma unroll
    for (int off = 16; off > 0; off >>= 1) sum += __shfl_xor(sum, off, 32);
    const float inv = __builtin_amdgcn_rcpf(sum);
    for (int j = lane; j < LK_; j += 32) {
      float p = row[j] * inv;
      row[j] = p;
      attn_base[(size_t)r * LK_ + j] = p;
    }
  }
  __syncthreads();

  // ---- Phase 3: O = P V. 8 waves: 4 d-tiles x 2 K-halves ----
  const int dtile = wid & 3;
  const int khalf = wid >> 2;
  const int d0 = dtile << 4;
  v8f acc = {};
  for (int kb0 = khalf * 1024; kb0 < khalf * 1024 + 1024; kb0 += 32) {
    v16bf ap, bv;
#pragma unroll
    for (int e = 0; e < 16; ++e) {
      int kk = kof(e, half);
      ap[e] = (__bf16)s_tile[l16 * SD + kb0 + kk];           // conflict-free (stride 2051)
      bv[e] = (__bf16)v[((size_t)b * LK_ + kb0 + kk) * D_ + d0 + l16];
    }
    acc = __builtin_amdgcn_wmma_f32_16x16x32_bf16(false, ap, false, bv, (short)0, acc, false, false);
  }
#pragma unroll
  for (int g = 0; g < 8; ++g) {
    const int m = g + 8 * half;
    o_scr[wid * 256 + m * 16 + l16] = acc[g];
  }
  __syncthreads();

  // ---- Phase 4: combine K-halves into a contiguous 16x64 tile, TDM-store it ----
  float vals[4];
#pragma unroll
  for (int rep = 0; rep < 4; ++rep) {
    const int lin = tid + rep * 256;       // 1024 outputs = 16 x 64
    const int m = lin >> 6;
    const int d = lin & 63;
    const int dt = d >> 4, n = d & 15;
    vals[rep] = o_scr[dt * 256 + m * 16 + n] + o_scr[(4 + dt) * 256 + m * 16 + n];
  }
  __syncthreads();
#pragma unroll
  for (int rep = 0; rep < 4; ++rep) o_scr[tid + rep * 256] = vals[rep];  // o_final[m*64+d]
  __syncthreads();

  if (wid == 0) {
    // Tensor Data Mover: store the contiguous 16x64 f32 tile (4KB) from LDS.
    // D# per CDNA5 ISA ch.8: group0 = {flags, lds_addr, global_addr, type=2},
    // group1 = {data_size=4B, tensor_dim0=1024, tensor_dim1=1, tile=1024x1, stride=1024}.
    unsigned long long ga =
        (unsigned long long)(uintptr_t)(out + ((size_t)b * LQ_ + q0) * (size_t)D_);
    u32x4 g0;
    g0[0] = 1u;                                   // count=1, user descriptor
    g0[1] = (unsigned)(16 * SD * 4);              // lds byte offset of o_final
    g0[2] = (unsigned)(ga & 0xFFFFFFFFu);         // global_addr[31:0]
    g0[3] = (unsigned)((ga >> 32) & 0x1FFFFFFu) | 0x80000000u;  // addr[56:32] | type=2
    i32x8 g1 = {};
    g1[0] = (int)(2u << 16);                      // data_size code 2 = 4 bytes
    g1[1] = (int)(1024u << 16);                   // tensor_dim0 = 1024 (bits 79:48, low)
    g1[2] = (int)(1u << 16);                      // tensor_dim0 hi=0 | tensor_dim1 = 1
    g1[3] = (int)(1024u << 16);                   // tile_dim0 = 1024 (bits 127:112)
    g1[4] = 1;                                    // tile_dim1 = 1
    g1[5] = 1024;                                 // tensor_dim0_stride = 1024
    i32x4 gz = {};
#if __clang_major__ >= 23
    i32x8 gz8 = {};
    __builtin_amdgcn_tensor_store_from_lds(g0, g1, gz, gz, gz8, 0);
#else
    __builtin_amdgcn_tensor_store_from_lds(g0, g1, gz, gz, 0);
#endif
    __builtin_amdgcn_s_wait_tensorcnt(0);
  }
}

extern "C" void kernel_launch(void* const* d_in, const int* in_sizes, int n_in,
                              void* d_out, int out_size, void* d_ws, size_t ws_size,
                              hipStream_t stream) {
  const float*         q    = (const float*)d_in[0];
  const float*         k    = (const float*)d_in[1];
  const float*         v    = (const float*)d_in[2];
  const float*         td   = (const float*)d_in[3];
  const unsigned char* mask = (const unsigned char*)d_in[4]; // bool tensor, 1B/elem
  const float*         tpw  = (const float*)d_in[5];
  const float*         tmw  = (const float*)d_in[6];

  float* out  = (float*)d_out;                              // [B,LQ,D] first
  float* attn = out + (size_t)B_ * LQ_ * D_;                // then [B,LQ,LK]

  float* psum = (float*)d_ws;
  float* pmax = psum + 1024;
  float* scal = pmax + 1024;

  const long long n = (long long)B_ * LQ_ * LK_;
  td_reduce_partial<<<1024, 256, 0, stream>>>(td, psum, pmax, n);
  td_finalize<<<1, 256, 0, stream>>>(psum, pmax, 1024, tpw, scal, (float)n);

  const size_t shmem = (size_t)(16 * SD + 8 * 256) * sizeof(float); // ~139.5 KB < 320 KB/WGP
  attn_fused<<<B_ * (LQ_ / 16), 256, shmem, stream>>>(q, k, v, td, mask, tmw, scal, out, attn);
}
